// EquivariantGNN_7275674599859
// MI455X (gfx1250) — matrix operand order repaired and checked by reading, not compiled, not run on hardware
//
#include <hip/hip_runtime.h>
#include <hip/hip_bf16.h>

// EquivariantGNN message passing for MI455X (gfx1250, wave32).
// 16 edges per wave = M dimension of v_wmma_f32_16x16x32_bf16.
// RBF features generated directly in WMMA A-fragment layout; weights
// pre-packed to bf16 B-fragment layout in LDS once per block.

#define WAVES 8
#define CUTOFF_F 10.0f
#define PI_F 3.14159265358979323846f

typedef __attribute__((ext_vector_type(16))) __bf16 v16bf;
typedef __attribute__((ext_vector_type(8)))  float  v8f;

union Frag { v16bf v; unsigned u[8]; };

// 16-bit A-matrix 16x32 layout: lane<16 -> K in {0..7,16..23}; lane>=16 -> K in {8..15,24..31}
__device__ __forceinline__ int kA_idx(int lane, int j) {
  return ((lane >> 4) << 3) + ((j >> 2) << 4) + ((j & 3) << 1);
}
// 16-bit B-matrix 32x16 layout: lane<16 -> K in {0..7,16..23}; lane>=16 -> K in {8..15,24..31} (2 per VGPR)
__device__ __forceinline__ int kB_idx(int lane, int j) {
  return ((lane >> 4) << 3) + ((j >> 2) << 4) + ((j & 3) << 1);
}

__device__ __forceinline__ unsigned pack_bf16(float a, float b) {
  union { __bf16 h[2]; unsigned u; } x;
  x.h[0] = (__bf16)a; x.h[1] = (__bf16)b;
  return x.u;
}

// fast sigmoid/silu: v_exp_f32 + v_rcp_f32 (no IEEE div ladder)
__device__ __forceinline__ float sigm(float x) {
  return __builtin_amdgcn_rcpf(1.0f + __expf(-x));
}
__device__ __forceinline__ float silu(float x) { return x * sigm(x); }

__device__ __forceinline__ v16bf load_bfrag(const unsigned* sB, int f, int lane) {
  Frag x;
  #pragma unroll
  for (int j = 0; j < 8; ++j) x.u[j] = sB[f * 256 + j * 32 + lane];
  return x.v;
}

// Fragment table in LDS:
//   f 0..1   : W1  [32x32], N-tiles 0..1
//   f 2..7   : W2  [32x96], N-tiles 0..5 (scalar 0..3, vec-filter 4, vec-gate 5)
//   f 8..27  : Wa1 [160x64], f = 8 + kchunk*4 + ntile
__global__ void __launch_bounds__(32 * WAVES)
gnn_edge_kernel(const float* __restrict__ scalars, const float* __restrict__ vectors,
                const float* __restrict__ edge_vec,
                const float* __restrict__ W1, const float* __restrict__ b1,
                const float* __restrict__ W2, const float* __restrict__ b2,
                const float* __restrict__ Wa1, const float* __restrict__ ba1,
                const float* __restrict__ Wa2, const float* __restrict__ ba2,
                const int* __restrict__ edge_index,
                float* __restrict__ outS, float* __restrict__ outV,
                int N, int E) {
  __shared__ unsigned sB[28 * 256];           // 28 KB: bf16-packed B fragments
  __shared__ float sStage[WAVES][16][32];     // 16 KB: per-wave D->A transpose stage
  __shared__ int   sSrc[WAVES][16];
  __shared__ int   sDst[WAVES][16];
  __shared__ float sAtt[WAVES][16];
  __shared__ float sSh[WAVES][16][3];

  const int tid  = threadIdx.x;
  const int lane = tid & 31;
  const int w    = tid >> 5;
  const int m    = lane & 15;                  // edge row within tile / column within D
  const int mr   = (lane >> 4) << 3;           // D-fragment row base for this half-wave

  // ---- Pre-pack weights into bf16 B-fragment layout (once per block) ----
  for (int f = w; f < 28; f += WAVES) {
    const float* M; int ld, k0, n0;
    if (f < 2)      { M = W1;  ld = 32; k0 = 0;               n0 = f * 16; }
    else if (f < 8) { M = W2;  ld = 96; k0 = 0;               n0 = (f - 2) * 16; }
    else            { int g = f - 8; M = Wa1; ld = 64; k0 = (g >> 2) * 32; n0 = (g & 3) * 16; }
    const int n = n0 + m;
    #pragma unroll
    for (int j = 0; j < 8; ++j) {
      const int k = k0 + kB_idx(lane, j);
      sB[f * 256 + j * 32 + lane] = pack_bf16(M[k * ld + n], M[(k + 1) * ld + n]);
    }
  }
  __syncthreads();

  const int tile  = blockIdx.x * WAVES + w;    // one 16-edge tile per wave
  const int tiles = (E + 15) / 16;
  if (tile >= tiles) return;                   // uniform per wave (after barrier)
  const int e0 = tile * 16;

  // ---- Hoist all per-lane bias/weight constants (overlap with geometry math) ----
  const float b1v0  = b1[m],        b1v1  = b1[16 + m];
  const float b2s0  = b2[m],        b2s1  = b2[16 + m];
  const float b2s2  = b2[32 + m],   b2s3  = b2[48 + m];
  const float b2vf  = b2[64 + m],   b2vg  = b2[80 + m];
  float ba1v[4], wa2v[4];
  #pragma unroll
  for (int t = 0; t < 4; ++t) { ba1v[t] = ba1[t * 16 + m]; wa2v[t] = Wa2[t * 16 + m]; }
  const float ba2v = ba2[0];

  // ---- Phase 0: per-edge geometry (both half-waves compute d for their row) ----
  const int  e     = e0 + m;
  const bool valid = (e < E);
  const int  ec    = valid ? e : 0;
  const int  srcn  = edge_index[ec];
  const int  dstn  = edge_index[E + ec];
  const float ex = edge_vec[ec * 3 + 0];
  const float ey = edge_vec[ec * 3 + 1];
  const float ez = edge_vec[ec * 3 + 2];
  const float d    = __builtin_sqrtf(ex * ex + ey * ey + ez * ez);
  const float invd = 1.0f / fmaxf(d, 1e-8f);
  float cut = (d < CUTOFF_F) ? 0.5f * (__cosf((PI_F / CUTOFF_F) * d) + 1.0f) : 0.0f;
  if (!valid) cut = 0.0f;                      // zero RBF rows of padding edges
  if (lane < 16) {
    sSrc[w][m] = srcn; sDst[w][m] = dstn;
    sSh[w][m][0] = ey * invd;                  // sh = (unit_y, unit_z, unit_x)
    sSh[w][m][1] = ez * invd;
    sSh[w][m][2] = ex * invd;
  }

  // ---- Phase 1: RBF directly in A-fragment layout ----
  Frag ra;
  const float wfreq = PI_F / CUTOFF_F;
  const float amp   = invd * cut;
  #pragma unroll
  for (int j = 0; j < 8; ++j) {
    const int k = kA_idx(lane, j);
    const float f0 = __sinf(wfreq * (float)(k + 1) * d) * amp;
    const float f1 = __sinf(wfreq * (float)(k + 2) * d) * amp;
    ra.u[j] = pack_bf16(f0, f1);
  }

  const v8f zero = {0.f, 0.f, 0.f, 0.f, 0.f, 0.f, 0.f, 0.f};

  // ---- Phase 2: h = silu(rbf @ W1 + b1), staged for re-layout ----
  #pragma unroll
  for (int t = 0; t < 2; ++t) {
    v16bf bf = load_bfrag(sB, t, lane);
    v8f dacc = __builtin_amdgcn_wmma_f32_16x16x32_bf16(false, ra.v, false, bf,
                                                       (short)0, zero, false, false);
    const float bias = (t == 0) ? b1v0 : b1v1;
    #pragma unroll
    for (int r = 0; r < 8; ++r)
      sStage[w][mr + r][t * 16 + m] = silu(dacc[r] + bias);
  }

  // ---- Phase 3: re-pack h into A-fragment layout (ds_load_b64 pairs) ----
  Frag ah;
  #pragma unroll
  for (int j = 0; j < 8; ++j) {
    const int k = kA_idx(lane, j);
    const float2 hp = *(const float2*)&sStage[w][m][k];
    ah.u[j] = pack_bf16(hp.x, hp.y);
  }

  // ---- Phase 4: attention  att_in[16x160] @ Wa1[160x64]  (5 K-chunks x 4 N-tiles) ----
  v8f acc[4] = {zero, zero, zero, zero};
  #pragma unroll
  for (int cc = 0; cc < 5; ++cc) {
    Frag aa;
    if (cc == 4) {
      aa = ra;                                 // rbf chunk reuses the live fragment
    } else {
      const int node = (cc < 2) ? sDst[w][m] : sSrc[w][m];   // s_i then s_j
      const float* srow = scalars + node * 64 + (cc & 1) * 32;
      #pragma unroll
      for (int j = 0; j < 8; ++j) {
        const float2 sp = *(const float2*)(srow + kA_idx(lane, j));  // global_load_b64
        aa.u[j] = pack_bf16(sp.x, sp.y);
      }
    }
    #pragma unroll
    for (int t = 0; t < 4; ++t) {
      v16bf bf = load_bfrag(sB, 8 + cc * 4 + t, lane);
      acc[t] = __builtin_amdgcn_wmma_f32_16x16x32_bf16(false, aa.v, false, bf,
                                                       (short)0, acc[t], false, false);
    }
  }
  // second layer: z = silu(hidden); dot with Wa2 -> shfl-butterfly over the 16 D-columns
  float p[8] = {0.f, 0.f, 0.f, 0.f, 0.f, 0.f, 0.f, 0.f};
  #pragma unroll
  for (int t = 0; t < 4; ++t) {
    #pragma unroll
    for (int r = 0; r < 8; ++r)
      p[r] += silu(acc[t][r] + ba1v[t]) * wa2v[t];
  }
  #pragma unroll
  for (int off = 1; off < 16; off <<= 1) {
    #pragma unroll
    for (int r = 0; r < 8; ++r) p[r] += __shfl_xor(p[r], off, 32);
  }
  if (m == 0) {
    #pragma unroll
    for (int r = 0; r < 8; ++r) sAtt[w][mr + r] = sigm(p[r] + ba2v);
  }

  // ---- Hoist per-row metadata for this half-wave's 8 D rows ----
  float attv[8], shv0[8], shv1[8], shv2[8];
  int   srcv[8], dstv[8];
  #pragma unroll
  for (int r = 0; r < 8; ++r) {
    const int mm = mr + r;
    attv[r] = sAtt[w][mm];
    srcv[r] = sSrc[w][mm];
    dstv[r] = sDst[w][mm];
    shv0[r] = sSh[w][mm][0];
    shv1[r] = sSh[w][mm][1];
    shv2[r] = sSh[w][mm][2];
  }

  // ---- Phase 5: filters = h @ W2 + b2, fused with message formation + scatter-add ----
  // scalar channels: tiles 0..3 (cols 0..63)
  const float b2sv[4] = {b2s0, b2s1, b2s2, b2s3};
  #pragma unroll
  for (int t = 0; t < 4; ++t) {
    v16bf bf = load_bfrag(sB, 2 + t, lane);
    v8f dv = __builtin_amdgcn_wmma_f32_16x16x32_bf16(false, ah.v, false, bf,
                                                     (short)0, zero, false, false);
    const int s = t * 16 + m;
    #pragma unroll
    for (int r = 0; r < 8; ++r) {
      if (e0 + mr + r < E) {
        const float filt = dv[r] + b2sv[t];
        const float msg  = attv[r] * scalars[srcv[r] * 64 + s] * filt;
        atomicAdd(&outS[dstv[r] * 64 + s], msg);
      }
    }
  }
  // vector channels: tile 4 = vector_filter (cols 64..79), tile 5 = vector_gate (80..95)
  {
    v16bf bf4 = load_bfrag(sB, 6, lane);
    v16bf bf5 = load_bfrag(sB, 7, lane);
    v8f dvf = __builtin_amdgcn_wmma_f32_16x16x32_bf16(false, ah.v, false, bf4,
                                                      (short)0, zero, false, false);
    v8f dvg = __builtin_amdgcn_wmma_f32_16x16x32_bf16(false, ah.v, false, bf5,
                                                      (short)0, zero, false, false);
    const int vi = m;                          // vector-channel index 0..15
    #pragma unroll
    for (int r = 0; r < 8; ++r) {
      if (e0 + mr + r < E) {
        const float vf = dvf[r] + b2vf;
        const float vg = dvg[r] + b2vg;
        const float a  = attv[r];
        const int vb = (srcv[r] * 16 + vi) * 3;
        const int ob = (dstv[r] * 16 + vi) * 3;
        const float sh[3] = {shv0[r], shv1[r], shv2[r]};
        #pragma unroll
        for (int c = 0; c < 3; ++c) {
          const float vm = a * (vectors[vb + c] * vf + vg * sh[c]);
          atomicAdd(&outV[ob + c], vm);
        }
      }
    }
  }
}

__global__ void init_out_kernel(const float* __restrict__ scalars,
                                const float* __restrict__ vectors,
                                float* __restrict__ out, int nS, int nV) {
  const int i = blockIdx.x * blockDim.x + threadIdx.x;
  if (i < nS)            out[i] = scalars[i];
  else if (i < nS + nV)  out[i] = vectors[i - nS];
}

extern "C" void kernel_launch(void* const* d_in, const int* in_sizes, int n_in,
                              void* d_out, int out_size, void* d_ws, size_t ws_size,
                              hipStream_t stream) {
  const float* scalars  = (const float*)d_in[0];
  const float* vectors  = (const float*)d_in[1];
  const float* edge_vec = (const float*)d_in[2];
  const float* W1  = (const float*)d_in[3];
  const float* b1  = (const float*)d_in[4];
  const float* W2  = (const float*)d_in[5];
  const float* b2  = (const float*)d_in[6];
  const float* Wa1 = (const float*)d_in[7];
  const float* ba1 = (const float*)d_in[8];
  const float* Wa2 = (const float*)d_in[9];
  const float* ba2 = (const float*)d_in[10];
  const int*   edge_index = (const int*)d_in[11];

  const int N = in_sizes[0] / 64;
  const int E = in_sizes[2] / 3;
  float* outS = (float*)d_out;
  float* outV = outS + N * 64;

  const int total = N * 64 + N * 48;
  init_out_kernel<<<(total + 255) / 256, 256, 0, stream>>>(scalars, vectors,
                                                           (float*)d_out, N * 64, N * 48);

  const int tiles  = (E + 15) / 16;
  const int blocks = (tiles + WAVES - 1) / WAVES;
  gnn_edge_kernel<<<blocks, 32 * WAVES, 0, stream>>>(scalars, vectors, edge_vec,
                                                     W1, b1, W2, b2, Wa1, ba1, Wa2, ba2,
                                                     edge_index, outS, outV, N, E);
}
